// LongRegression_81338090652108
// MI455X (gfx1250) — matrix operand order repaired and checked by reading, not compiled, not run on hardware
//
#include <hip/hip_runtime.h>
#include <hip/hip_bf16.h>

// ---------------------------------------------------------------------------
// Soft-MoE fused forward for MI455X (gfx1250, wave32, WMMA bf16 path).
//   x:[16384,512] -> router gates [16384,8]; 8 experts 512->1024->1024->256;
//   out[n,o] = sum_e gates[n,e] * expert_e(x)[n,o]
//
// Round 3: register-blocked WMMA. Each wave computes a 2(M)x4(N) tile block
// with K outermost: per 32-wide K step, 2 A-fragments (LDS) + 4 B-fragments
// (global bf16) feed 8 v_wmma_f32_16x16x32_bf16 -> 0.75 mem ops per WMMA,
// 4x less LDS A-traffic, 2x less L2 B-traffic than the unblocked version.
// ---------------------------------------------------------------------------

typedef __attribute__((ext_vector_type(16))) __bf16 bf16x16;
typedef __attribute__((ext_vector_type(8)))  __bf16 bf16x8;
typedef __attribute__((ext_vector_type(8)))  float  f32x8;

#define TM 32           // tokens per workgroup tile (2 M-tiles of 16)
#define NTHREADS 256    // 8 wave32 per WG

// ---- fp32 -> bf16 bulk conversion (weights), vectorized, grid-stride
__global__ __launch_bounds__(NTHREADS)
void cvt_f32_bf16_kernel(const float* __restrict__ src, __bf16* __restrict__ dst,
                         int n4 /* count/4 */) {
    int i = blockIdx.x * blockDim.x + threadIdx.x;
    int stride = gridDim.x * blockDim.x;
    for (; i < n4; i += stride) {
        float4 f = *(const float4*)(src + (size_t)i * 4);
        __bf16* d = dst + (size_t)i * 4;
        d[0] = (__bf16)f.x; d[1] = (__bf16)f.y;
        d[2] = (__bf16)f.z; d[3] = (__bf16)f.w;
    }
}

// ---- A fragment: 16x32 bf16 tile from LDS (activations), ISA 7.12.2 layout.
// lanes 0-15: M=lane, VGPR0-3 hold K=0..7 (pairs), VGPR4-7 hold K=16..23
// lanes16-31: M=lane-16, VGPR0-3 hold K=8..15,     VGPR4-7 hold K=24..31
__device__ __forceinline__ bf16x16
load_a_lds(const __bf16* __restrict__ tile, int rs, int m0, int k0, int lane) {
    const int m  = m0 + (lane & 15);
    const int kb = k0 + ((lane & 16) ? 8 : 0);
    const __bf16* p = tile + m * rs + kb;     // 16B-aligned (k0 % 32 == 0)
    bf16x8 lo = *(const bf16x8*)(p);          // K = kb .. kb+7
    bf16x8 hi = *(const bf16x8*)(p + 16);     // K = kb+16 .. kb+23
    bf16x16 a;
#pragma unroll
    for (int j = 0; j < 8; ++j) { a[j] = lo[j]; a[j + 8] = hi[j]; }
    return a;
}

// ---- B fragment: 32x16 bf16 tile from global pre-converted weights.
// lane: N column n0+lane%16; elements 0-7 K=kb..kb+7, 8-15 K=kb+16..kb+23 (kb=k0+8*(lane/16)).
__device__ __forceinline__ bf16x16
load_b_global(const __bf16* __restrict__ W, int ldb, int k0, int n0, int lane) {
    const int kb = k0 + ((lane & 16) ? 8 : 0);
    const __bf16* p = W + (size_t)kb * ldb + n0 + (lane & 15);
    __builtin_prefetch(p + (size_t)32 * ldb, 0, 1);   // next K-slab -> global_prefetch_b8
    bf16x16 b;
#pragma unroll
    for (int j = 0; j < 8; ++j) { b[j] = p[(size_t)j * ldb]; b[j + 8] = p[(size_t)(16 + j) * ldb]; }
    return b;
}

// ---- register-blocked GEMM: one wave computes a 2(M) x NT(N) block of
// 16x16 fp32 tiles. K outer loop: 2 A-frags + NT B-frags -> 2*NT WMMAs.
template<int NT>
__device__ __forceinline__ void
wmma_block(const __bf16* __restrict__ As, int rs,
           const __bf16* __restrict__ W, int ldb, int n0, int K, int lane,
           f32x8 (&acc)[2][NT]) {
    for (int k0 = 0; k0 < K; k0 += 32) {
        bf16x16 a0 = load_a_lds(As, rs, 0,  k0, lane);
        bf16x16 a1 = load_a_lds(As, rs, 16, k0, lane);
#pragma unroll
        for (int j = 0; j < NT; ++j) {
            bf16x16 b = load_b_global(W, ldb, k0, n0 + 16 * j, lane);
            acc[0][j] = __builtin_amdgcn_wmma_f32_16x16x32_bf16(
                false, a0, false, b, (short)0, acc[0][j], false, false);
            acc[1][j] = __builtin_amdgcn_wmma_f32_16x16x32_bf16(
                false, a1, false, b, (short)0, acc[1][j], false, false);
        }
    }
}

// ---- epilogue: bias + relu, store bf16 to LDS. C layout: VGPR r ->
// lanes 0-15: M=r, N=lane; lanes 16-31: M=8+r, N=lane-16.
__device__ __forceinline__ void
store_tile_relu(__bf16* __restrict__ Hs, int rs, int m0, int n0,
                const float* __restrict__ bias, f32x8 acc, int lane) {
    const int n     = lane & 15;
    const int mhalf = (lane & 16) ? 8 : 0;
    const float bv  = bias[n0 + n];
#pragma unroll
    for (int r = 0; r < 8; ++r) {
        float v = acc[r] + bv;
        v = v > 0.0f ? v : 0.0f;
        Hs[(m0 + mhalf + r) * rs + n0 + n] = (__bf16)v;
    }
}

// ===========================================================================
// Router: gates[N,8] = softmax(relu(relu(x@r1+rb1)@r2+rb2)@rout+rbout)
// dyn LDS: Xs 32KB (aliased by H2r 16KB after L1) | H1r 16KB | logits 1KB
// ===========================================================================
__global__ __launch_bounds__(NTHREADS)
void moe_router_kernel(const float* __restrict__ x,
                       const __bf16* __restrict__ r1b, const float* __restrict__ rb1,
                       const __bf16* __restrict__ r2b, const float* __restrict__ rb2,
                       const float* __restrict__ rout, const float* __restrict__ rbout,
                       float* __restrict__ gates) {
    extern __shared__ char smem[];
    __bf16* Xs     = (__bf16*)smem;              // [32][512]
    __bf16* H2r    = (__bf16*)smem;              // [32][256] (aliases Xs)
    __bf16* H1r    = (__bf16*)(smem + 32768);    // [32][256]
    float*  logits = (float*)(smem + 32768 + 16384); // [32][8]

    const int tid  = threadIdx.x;
    const int lane = tid & 31;
    const int wid  = tid >> 5;
    const int tok0 = blockIdx.x * TM;

    // stage X tile fp32 -> bf16 (coalesced float4)
    const float* xsrc = x + (size_t)tok0 * 512;
#pragma unroll
    for (int i = 0; i < 16; ++i) {
        int idx = (tid + i * NTHREADS) * 4;
        float4 f = *(const float4*)(xsrc + idx);
        __bf16* d = Xs + idx;
        d[0] = (__bf16)f.x; d[1] = (__bf16)f.y;
        d[2] = (__bf16)f.z; d[3] = (__bf16)f.w;
    }
    __syncthreads();

    // L1: [32x256] = relu(Xs @ r1 + rb1), K=512. 2x2 block per wave.
    {
        const int n0 = wid * 2 * 16;
        f32x8 acc[2][2];
#pragma unroll
        for (int mi = 0; mi < 2; ++mi)
#pragma unroll
            for (int j = 0; j < 2; ++j) acc[mi][j] = (f32x8)0.0f;
        wmma_block<2>(Xs, 512, r1b, 256, n0, 512, lane, acc);
#pragma unroll
        for (int j = 0; j < 2; ++j) {
            store_tile_relu(H1r, 256, 0,  n0 + 16 * j, rb1, acc[0][j], lane);
            store_tile_relu(H1r, 256, 16, n0 + 16 * j, rb1, acc[1][j], lane);
        }
    }
    __syncthreads();

    // L2: [32x256] = relu(H1r @ r2 + rb2), K=256  (H2r aliases Xs: Xs dead)
    {
        const int n0 = wid * 2 * 16;
        f32x8 acc[2][2];
#pragma unroll
        for (int mi = 0; mi < 2; ++mi)
#pragma unroll
            for (int j = 0; j < 2; ++j) acc[mi][j] = (f32x8)0.0f;
        wmma_block<2>(H1r, 256, r2b, 256, n0, 256, lane, acc);
#pragma unroll
        for (int j = 0; j < 2; ++j) {
            store_tile_relu(H2r, 256, 0,  n0 + 16 * j, rb2, acc[0][j], lane);
            store_tile_relu(H2r, 256, 16, n0 + 16 * j, rb2, acc[1][j], lane);
        }
    }
    __syncthreads();

    // logits: one (token, expert) pair per thread (E=8 too small for WMMA)
    {
        const int t = tid >> 3, e = tid & 7;
        float s = rbout[e];
        const __bf16* h = H2r + t * 256;
        for (int k = 0; k < 256; ++k) s += (float)h[k] * rout[k * 8 + e];
        logits[t * 8 + e] = s;
    }
    __syncthreads();

    // softmax over 8 experts, one token per thread
    if (tid < TM) {
        float l[8], m = -1e30f, s = 0.0f;
#pragma unroll
        for (int e = 0; e < 8; ++e) { l[e] = logits[tid * 8 + e]; m = fmaxf(m, l[e]); }
#pragma unroll
        for (int e = 0; e < 8; ++e) { l[e] = expf(l[e] - m); s += l[e]; }
        const float inv = 1.0f / s;
#pragma unroll
        for (int e = 0; e < 8; ++e)
            gates[(size_t)(tok0 + tid) * 8 + e] = l[e] * inv;
    }
}

// ===========================================================================
// Experts, fully fused over e with gate-combine in registers.
// dyn LDS: Xs 32KB | H1s 64KB | H2s 64KB | gates 1KB  = 161.0 KB (< 320 KB WGP)
// ===========================================================================
__global__ __launch_bounds__(NTHREADS)
void moe_expert_kernel(const float* __restrict__ x,
                       const __bf16* __restrict__ w1b, const float* __restrict__ b1,
                       const __bf16* __restrict__ w2b, const float* __restrict__ b2,
                       const __bf16* __restrict__ woutb, const float* __restrict__ bout,
                       const float* __restrict__ gates, float* __restrict__ out) {
    extern __shared__ char smem[];
    __bf16* Xs  = (__bf16*)smem;                     // [32][512]
    __bf16* H1s = (__bf16*)(smem + 32768);           // [32][1024]
    __bf16* H2s = (__bf16*)(smem + 32768 + 65536);   // [32][1024]
    float*  gs  = (float*)(smem + 32768 + 131072);   // [32][8]

    const int tid  = threadIdx.x;
    const int lane = tid & 31;
    const int wid  = tid >> 5;
    const int tok0 = blockIdx.x * TM;

    // stage X tile fp32 -> bf16 and this tile's gates
    const float* xsrc = x + (size_t)tok0 * 512;
#pragma unroll
    for (int i = 0; i < 16; ++i) {
        int idx = (tid + i * NTHREADS) * 4;
        float4 f = *(const float4*)(xsrc + idx);
        __bf16* d = Xs + idx;
        d[0] = (__bf16)f.x; d[1] = (__bf16)f.y;
        d[2] = (__bf16)f.z; d[3] = (__bf16)f.w;
    }
    gs[tid] = gates[(size_t)tok0 * 8 + tid];         // 32*8 == 256 threads
    __syncthreads();

    f32x8 outAcc[2][2];                              // persistent combine regs
#pragma unroll
    for (int mi = 0; mi < 2; ++mi)
#pragma unroll
        for (int j = 0; j < 2; ++j) outAcc[mi][j] = (f32x8)0.0f;

    for (int e = 0; e < 8; ++e) {
        const __bf16* W1e = w1b   + (size_t)e * 512 * 1024;
        const float*  b1e = b1    + (size_t)e * 1024;
        const __bf16* W2e = w2b   + (size_t)e * 1024 * 1024;
        const float*  b2e = b2    + (size_t)e * 1024;
        const __bf16* Woe = woutb + (size_t)e * 1024 * 256;
        const float*  boe = bout  + (size_t)e * 256;

        // L1: H1[32x1024] = relu(Xs @ W1e + b1e), K=512.
        // 2x64 tiles; per wave: 2 chunks of a 2(M)x4(N) register block.
        for (int c = 0; c < 2; ++c) {
            const int n0 = (wid * 8 + c * 4) * 16;
            f32x8 acc[2][4];
#pragma unroll
            for (int mi = 0; mi < 2; ++mi)
#pragma unroll
                for (int j = 0; j < 4; ++j) acc[mi][j] = (f32x8)0.0f;
            wmma_block<4>(Xs, 512, W1e, 1024, n0, 512, lane, acc);
#pragma unroll
            for (int j = 0; j < 4; ++j) {
                store_tile_relu(H1s, 1024, 0,  n0 + 16 * j, b1e, acc[0][j], lane);
                store_tile_relu(H1s, 1024, 16, n0 + 16 * j, b1e, acc[1][j], lane);
            }
        }
        __syncthreads();

        // L2: H2[32x1024] = relu(H1 @ W2e + b2e), K=1024. Same blocking.
        for (int c = 0; c < 2; ++c) {
            const int n0 = (wid * 8 + c * 4) * 16;
            f32x8 acc[2][4];
#pragma unroll
            for (int mi = 0; mi < 2; ++mi)
#pragma unroll
                for (int j = 0; j < 4; ++j) acc[mi][j] = (f32x8)0.0f;
            wmma_block<4>(H1s, 1024, W2e, 1024, n0, 1024, lane, acc);
#pragma unroll
            for (int j = 0; j < 4; ++j) {
                store_tile_relu(H2s, 1024, 0,  n0 + 16 * j, b2e, acc[0][j], lane);
                store_tile_relu(H2s, 1024, 16, n0 + 16 * j, b2e, acc[1][j], lane);
            }
        }
        __syncthreads();

        // L3 + combine: out += gate[n,e] * (H2 @ Woe + boe), K=1024.
        // 2x16 tiles; per wave one 2(M)x2(N) block.
        {
            const int n0 = wid * 2 * 16;
            f32x8 acc[2][2];
#pragma unroll
            for (int mi = 0; mi < 2; ++mi)
#pragma unroll
                for (int j = 0; j < 2; ++j) acc[mi][j] = (f32x8)0.0f;
            wmma_block<2>(H2s, 1024, Woe, 256, n0, 1024, lane, acc);

            const int n = lane & 15, mhalf = (lane & 16) ? 8 : 0;
#pragma unroll
            for (int j = 0; j < 2; ++j) {
                const float bv = boe[n0 + 16 * j + n];
#pragma unroll
                for (int mi = 0; mi < 2; ++mi) {
#pragma unroll
                    for (int r = 0; r < 8; ++r) {
                        const int trow = mi * 16 + mhalf + r;
                        outAcc[mi][j][r] += gs[trow * 8 + e] * (acc[mi][j][r] + bv);
                    }
                }
            }
        }
        __syncthreads();   // protect H1s/H2s for next expert
    }

    // write fp32 output (each element written exactly once; no atomics)
    {
        const int n = lane & 15, mhalf = (lane & 16) ? 8 : 0;
        const int n0 = wid * 2 * 16;
#pragma unroll
        for (int j = 0; j < 2; ++j)
#pragma unroll
            for (int mi = 0; mi < 2; ++mi)
#pragma unroll
                for (int r = 0; r < 8; ++r) {
                    const int trow = mi * 16 + mhalf + r;
                    out[(size_t)(tok0 + trow) * 256 + n0 + 16 * j + n] =
                        outAcc[mi][j][r];
                }
    }
}

// ===========================================================================
extern "C" void kernel_launch(void* const* d_in, const int* in_sizes, int n_in,
                              void* d_out, int out_size, void* d_ws, size_t ws_size,
                              hipStream_t stream) {
    const float* x    = (const float*)d_in[0];
    const float* w1   = (const float*)d_in[1];
    const float* b1   = (const float*)d_in[2];
    const float* w2   = (const float*)d_in[3];
    const float* b2   = (const float*)d_in[4];
    const float* wout = (const float*)d_in[5];
    const float* bout = (const float*)d_in[6];
    const float* r1   = (const float*)d_in[7];
    const float* rb1  = (const float*)d_in[8];
    const float* r2   = (const float*)d_in[9];
    const float* rb2  = (const float*)d_in[10];
    const float* rout = (const float*)d_in[11];
    const float* rbout= (const float*)d_in[12];

    // ---- workspace layout (bytes) ----
    //   gates fp32 [16384,8]          @ 0        (512 KB)
    //   w1b  bf16 [8,512,1024]        @ 0x080000 (8 MB)
    //   w2b  bf16 [8,1024,1024]       @ 0x880000 (16 MB)
    //   woutb bf16 [8,1024,256]       @ 0x1880000 (4 MB)
    //   r1b  bf16 [512,256]           @ 0x1C80000 (256 KB)
    //   r2b  bf16 [256,256]           @ 0x1CC0000 (128 KB)
    //   total ~28.9 MB
    char* ws = (char*)d_ws;
    float*  gates = (float*)ws;
    __bf16* w1b   = (__bf16*)(ws + 0x080000);
    __bf16* w2b   = (__bf16*)(ws + 0x880000);
    __bf16* woutb = (__bf16*)(ws + 0x1880000);
    __bf16* r1b   = (__bf16*)(ws + 0x1C80000);
    __bf16* r2b   = (__bf16*)(ws + 0x1CC0000);

    float* out = (float*)d_out;        // [16384,256] fp32

    // ---- weight pre-conversion fp32 -> bf16 (once per call, tiny) ----
    {
        struct Seg { const float* s; __bf16* d; int n; };
        const Seg segs[5] = {
            { w1,   w1b,   8 * 512 * 1024 },
            { w2,   w2b,   8 * 1024 * 1024 },
            { wout, woutb, 8 * 1024 * 256 },
            { r1,   r1b,   512 * 256 },
            { r2,   r2b,   256 * 256 },
        };
        for (int s = 0; s < 5; ++s) {
            int n4 = segs[s].n / 4;
            int blocks = (n4 + NTHREADS - 1) / NTHREADS;
            if (blocks > 1024) blocks = 1024;
            cvt_f32_bf16_kernel<<<blocks, NTHREADS, 0, stream>>>(segs[s].s, segs[s].d, n4);
        }
    }

    const int nblocks = 16384 / TM;    // 512
    const size_t smem_router = 32768 + 16384 + 1024;          // 50,176 B
    const size_t smem_expert = 32768 + 65536 + 65536 + 1024;  // 164,864 B

    (void)hipFuncSetAttribute((const void*)moe_router_kernel,
                              hipFuncAttributeMaxDynamicSharedMemorySize,
                              (int)smem_router);
    (void)hipFuncSetAttribute((const void*)moe_expert_kernel,
                              hipFuncAttributeMaxDynamicSharedMemorySize,
                              (int)smem_expert);

    moe_router_kernel<<<nblocks, NTHREADS, smem_router, stream>>>(
        x, r1b, rb1, r2b, rb2, rout, rbout, gates);

    moe_expert_kernel<<<nblocks, NTHREADS, smem_expert, stream>>>(
        x, w1b, b1, w2b, b2, woutb, bout, gates, out);
}
